// CustomGRU_13443247636903
// MI455X (gfx1250) — matrix operand order repaired and checked by reading, not compiled, not run on hardware
//
#include <hip/hip_runtime.h>
#include <hip/hip_bf16.h>

// ---------------------------------------------------------------------------
// GRU on MI455X (gfx1250, wave32, WMMA).
//
// Batch-tiled recurrence: one wave owns 16 batch rows for all 512 timesteps.
// Per step: 18 v_wmma_f32_16x16x32_f16 (A = [x_t | h | 1], K pad 41->64; B =
// padded 41x48 gate weights + bias row at K=56, held in VGPRs), f32 accum.
// Gate order r -> z -> candidate shortens the r->rh->candidate critical path
// and keeps only one gate's accumulators live at a time (VGPR pressure).
// Activations use native v_tanh_f32; x loads pipelined one iteration ahead.
// ---------------------------------------------------------------------------

typedef __attribute__((ext_vector_type(16))) _Float16 v16h;
typedef __attribute__((ext_vector_type(8)))  _Float16 v8h;
typedef __attribute__((ext_vector_type(8)))  float    v8f;

#define GRU_B   8192
#define GRU_T   512
#define GRU_I   8
#define GRU_H   33
#define GRU_HOR 24
#define KTOT    (GRU_I + GRU_H)   // 41 combined K
#define KONE    56                // K row carrying the bias (A column pinned to 1.0)
#define STRIDE  72                // LDS row stride (halves); kills bank conflicts

#define FRAG_HALVES (18 * 32 * 16)

// ---------------------------------------------------------------------------
// Prep: swizzle f32 weights into per-lane WMMA B-fragment layout (f16), pad,
// and plant the bias vector in row KONE.
// B layout used here (contiguous k, same as A): lane L -> N = L%16, half j -> K = (L/16)*16 + j.
// Gate order in fragments: g=0 -> Wr, g=1 -> Wz, g=2 -> Wh (r first).
// ---------------------------------------------------------------------------
__global__ __launch_bounds__(256)
void gru_prep_kernel(const float* __restrict__ Wz, const float* __restrict__ bz,
                     const float* __restrict__ Wr, const float* __restrict__ br,
                     const float* __restrict__ Wh, const float* __restrict__ bh,
                     _Float16* __restrict__ frag)
{
    const float* Ws[3] = {Wr, Wz, Wh};
    const float* bs[3] = {br, bz, bh};
    for (int idx = threadIdx.x; idx < FRAG_HALVES; idx += blockDim.x) {
        int j  = idx & 15;
        int l  = (idx >> 4) & 31;
        int f  = idx >> 9;           // 0..17
        int kc = f & 1;
        int nt = (f >> 1) % 3;
        int g  = f / 6;
        int K  = kc * 32 + (l >> 4) * 16 + j;
        int N  = nt * 16 + (l & 15);
        float v = 0.0f;
        if (N < GRU_H) {
            if (K < KTOT)       v = Ws[g][K * GRU_H + N];
            else if (K == KONE) v = bs[g][N];     // bias row, times A's 1-column
        }
        frag[idx] = (_Float16)v;
    }
}

// ---- fast activations on native TRANS32 ops ----
#if __has_builtin(__builtin_amdgcn_tanhf)
__device__ __forceinline__ float fast_tanh(float x) { return __builtin_amdgcn_tanhf(x); }
__device__ __forceinline__ float fast_sigmoid(float x) {
    return __builtin_fmaf(0.5f, __builtin_amdgcn_tanhf(0.5f * x), 0.5f);
}
#else
__device__ __forceinline__ float fast_tanh(float x) {
    float e = __builtin_amdgcn_exp2f(x * -2.885390082f);
    return (1.0f - e) * __builtin_amdgcn_rcpf(1.0f + e);
}
__device__ __forceinline__ float fast_sigmoid(float x) {
    return __builtin_amdgcn_rcpf(1.0f + __builtin_amdgcn_exp2f(x * -1.442695041f));
}
#endif

// ---------------------------------------------------------------------------
// Main: one wave per 16-batch tile, sequential over T.
// A layout used here (contiguous k, same as B): lane L -> M = L%16, half j -> K = (L/16)*16 + j.
// C/D layout (16x16 f32): lane L -> N = L%16, reg r -> M = r + (L/16)*8.
// Stage columns: 0..7 = x_t, 8..55 = h / r*h (48 padded), 56 = constant 1.0,
// 57..71 = zero pad. Loop stores never touch col 56.
// ---------------------------------------------------------------------------
__global__ __launch_bounds__(32)
void gru_main_kernel(const float* __restrict__ x,
                     const _Float16* __restrict__ frag,
                     const float* __restrict__ Wfc,
                     const float* __restrict__ bfc,
                     float* __restrict__ out)
{
    __shared__ _Float16 stage[16 * STRIDE];   // [x(8) | h(48) | 1 | pad] per row, f16
    __shared__ float    hfin[16 * GRU_H];

    const int lane = threadIdx.x;   // 0..31, single wave
    const int ln   = lane & 15;
    const int hi   = lane >> 4;
    const int b0   = blockIdx.x * 16;

    // zero staging (pads + h start at 0), then pin the bias-ones column
    for (int i = lane; i < 16 * STRIDE; i += 32) stage[i] = (_Float16)0.0f;
    if (lane < 16) stage[lane * STRIDE + KONE] = (_Float16)1.0f;

    // preload all 18 B fragments into VGPRs (g: 0=r, 1=z, 2=h~)
    v16h Bf[3][3][2];
    #pragma unroll
    for (int g = 0; g < 3; ++g)
        #pragma unroll
        for (int nt = 0; nt < 3; ++nt)
            #pragma unroll
            for (int kc = 0; kc < 2; ++kc) {
                int f = (g * 3 + nt) * 2 + kc;
                Bf[g][nt][kc] = *(const v16h*)(frag + f * 512 + lane * 16);
            }

    const v8f vzero = {0.f, 0.f, 0.f, 0.f, 0.f, 0.f, 0.f, 0.f};
    v8f hc[3] = {vzero, vzero, vzero};   // h in C-layout, f32

    const int xr = lane >> 1, xc = (lane & 1) * 4;
    const float* xrowp = x + (size_t)(b0 + xr) * (GRU_T * GRU_I) + xc;

    __syncthreads();

    // software pipeline: xv holds x_t while the load for x_{t+1} is in flight
    float4 xv = *(const float4*)(xrowp);

    #pragma unroll 1
    for (int t = 0; t < GRU_T; ++t) {
        // ---- stage x_t into columns 0..7 (f32 -> f16) ----
        _Float16* sp = &stage[xr * STRIDE + xc];
        sp[0] = (_Float16)xv.x; sp[1] = (_Float16)xv.y;
        sp[2] = (_Float16)xv.z; sp[3] = (_Float16)xv.w;
        // prefetch next timestep's x (full iteration to cover latency)
        {
            int tn = (t + 1 < GRU_T) ? (t + 1) : t;
            xv = *(const float4*)(xrowp + tn * GRU_I);
        }
        __syncthreads();

        // ---- gather A = [x_t | h | 1] (two K-chunks) ----
        union AU { v16h v; v8h h[2]; };
        AU a0, a1;
        {
            const int base = ln * STRIDE + hi * 16;
            a0.h[0] = *(const v8h*)&stage[base];
            a0.h[1] = *(const v8h*)&stage[base + 8];
            a1.h[0] = *(const v8h*)&stage[base + 32];
            a1.h[1] = *(const v8h*)&stage[base + 40];
        }

        // ---- r gate first (critical path into candidate GEMM) ----
        {
            v8f rc[3];
            #pragma unroll
            for (int nt = 0; nt < 3; ++nt) {
                v8f c = vzero;
                c = __builtin_amdgcn_wmma_f32_16x16x32_f16(false, a0.v, false, Bf[0][nt][0], (short)0, c, false, false);
                c = __builtin_amdgcn_wmma_f32_16x16x32_f16(false, a1.v, false, Bf[0][nt][1], (short)0, c, false, false);
                rc[nt] = c;
            }
            #pragma unroll
            for (int nt = 0; nt < 3; ++nt)
                #pragma unroll
                for (int k = 0; k < 8; ++k) {
                    float rg = fast_sigmoid(rc[nt][k]);
                    float rh = rg * hc[nt][k];   // pad cols: 0.5 * 0 = 0, stays clean
                    stage[(k + hi * 8) * STRIDE + GRU_I + nt * 16 + ln] = (_Float16)rh;
                }
        }

        // ---- z gate WMMAs (overlap with candidate phase; rc regs retired) ----
        v8f zc[3];
        #pragma unroll
        for (int nt = 0; nt < 3; ++nt) {
            v8f c = vzero;
            c = __builtin_amdgcn_wmma_f32_16x16x32_f16(false, a0.v, false, Bf[1][nt][0], (short)0, c, false, false);
            c = __builtin_amdgcn_wmma_f32_16x16x32_f16(false, a1.v, false, Bf[1][nt][1], (short)0, c, false, false);
            zc[nt] = c;
        }
        __syncthreads();

        // ---- candidate gate: A = [x_t | r*h | 1]; fused update ----
        AU c0, c1;
        {
            const int base = ln * STRIDE + hi * 16;
            c0.h[0] = *(const v8h*)&stage[base];
            c0.h[1] = *(const v8h*)&stage[base + 8];
            c1.h[0] = *(const v8h*)&stage[base + 32];
            c1.h[1] = *(const v8h*)&stage[base + 40];
        }
        #pragma unroll
        for (int nt = 0; nt < 3; ++nt) {
            v8f c = vzero;
            c = __builtin_amdgcn_wmma_f32_16x16x32_f16(false, c0.v, false, Bf[2][nt][0], (short)0, c, false, false);
            c = __builtin_amdgcn_wmma_f32_16x16x32_f16(false, c1.v, false, Bf[2][nt][1], (short)0, c, false, false);
            #pragma unroll
            for (int k = 0; k < 8; ++k) {
                float z  = fast_sigmoid(zc[nt][k]);            // deferred z-sigmoid
                float ht = fast_tanh(c[k]);
                float hn = __builtin_fmaf(z, ht - hc[nt][k], hc[nt][k]); // h + z*(h~-h)
                hc[nt][k] = hn;
                stage[(k + hi * 8) * STRIDE + GRU_I + nt * 16 + ln] = (_Float16)hn;
            }
        }
        __syncthreads();
    }

    // ---- final projection: out = h_final @ Wfc + bfc (16x33 @ 33x24) ----
    #pragma unroll
    for (int nt = 0; nt < 3; ++nt) {
        int n = nt * 16 + ln;
        if (n < GRU_H) {
            #pragma unroll
            for (int k = 0; k < 8; ++k)
                hfin[(k + hi * 8) * GRU_H + n] = hc[nt][k];
        }
    }
    __syncthreads();

    const int m  = ln;
    const int ob = hi * 12;       // lanes 0-15: outputs 0..11, lanes 16-31: 12..23
    #pragma unroll 1
    for (int o = 0; o < 12; ++o) {
        float acc = bfc[ob + o];
        for (int n = 0; n < GRU_H; ++n)
            acc += hfin[m * GRU_H + n] * Wfc[n * GRU_HOR + ob + o];
        out[(size_t)(b0 + m) * GRU_HOR + ob + o] = acc;
    }
}

// ---------------------------------------------------------------------------
extern "C" void kernel_launch(void* const* d_in, const int* in_sizes, int n_in,
                              void* d_out, int out_size, void* d_ws, size_t ws_size,
                              hipStream_t stream)
{
    const float* x   = (const float*)d_in[0];
    const float* Wz  = (const float*)d_in[1];
    const float* bz  = (const float*)d_in[2];
    const float* Wr  = (const float*)d_in[3];
    const float* br  = (const float*)d_in[4];
    const float* Wh  = (const float*)d_in[5];
    const float* bh  = (const float*)d_in[6];
    const float* Wfc = (const float*)d_in[7];
    const float* bfc = (const float*)d_in[8];
    float* out = (float*)d_out;

    _Float16* frag = (_Float16*)d_ws;

    gru_prep_kernel<<<1, 256, 0, stream>>>(Wz, bz, Wr, br, Wh, bh, frag);
    gru_main_kernel<<<GRU_B / 16, 32, 0, stream>>>(x, frag, Wfc, bfc, out);
}
